// RNNModel_57793079935440
// MI455X (gfx1250) — matrix operand order repaired and checked by reading, compiled, not run on hardware
//
#include <hip/hip_runtime.h>

// ---------------------------------------------------------------------------
// AWD-LSTM LM forward on MI455X (gfx1250, wave32, WMMA bf16).
// bf16 weights/activations (zero-padded), f32 accumulation via
// v_wmma_f32_16x16x32_bf16. Input projection fused into the recurrent GEMM via
// a split-K A operand: gates = [x_t | h] @ [Wih;Whh]^T + (b_ih+b_hh).
// GEMM template <WAVES, NTILES>: block tile (WAVES*16) x (NTILES*16); B panel
// staged through double-buffered LDS shared by all waves.
//   recurrence: <2,4>  -> 32x64 tiles, 288 blocks/step (occupancy on the
//                         serial critical path)
//   decoder:    <8,8>  -> 128x128 tiles, ~16 TB/s L2 demand at compute ceiling
// ---------------------------------------------------------------------------

typedef __bf16 bf16;
typedef __attribute__((ext_vector_type(16))) __bf16 v16bf;
typedef __attribute__((ext_vector_type(8)))  __bf16 v8bf;
typedef __attribute__((ext_vector_type(8)))  float  v8f;

#define TT 70
#define BB 128
#define EE 400
#define HH 1150
#define VV 33278
#define EP 416      // E padded to mult of 32
#define HP 1152     // H padded to mult of 32
#define G0P 4608    // 4H padded to mult of 64
#define G2P 1600    // 4E (already mult of 64)
#define VP 33280    // V padded to mult of 128
#define MR (TT*BB)  // 8960 rows (mult of 128)

#define LDSROW 40   // bf16 elems per LDS B row: 32 data + 8 pad (bank spread)

union Frag { v16bf v; v8bf h[2]; };

// ---------------------------------------------------------------------------
// NT GEMM: out[M,N] = A[M,K] * B[N,K]^T (+bias).  A split at column K1:
// cols [0,K1) from A1 (ld lda1), cols [K1,K) from A2 (ld lda2).
// Grid: (M/(WAVES*16), Npad/(NTILES*16)), block WAVES*32 threads. Wave w owns
// rows [blk*BM + w*16, +16); all waves share the LDS-staged (NTILES*16)x32 B
// tile (double buffered, one barrier per k-step).
// VGPR layouts per CDNA5 ISA:
//   A 16x32 bf16: lane m<16 holds K={k0..k0+7, k0+16..k0+23}; lanes 16-31 +8.
//   B 32x16 bf16: lane n (0-15) holds K={k0..k0+7, k0+16..k0+23}; lanes 16-31 +8.
//   C 16x16 f32:  lanes 0-15: N=lane, vgpr r -> M=r; lanes 16-31: M=r+8.
// ---------------------------------------------------------------------------
template<int WAVES, int NTILES>
__global__ __launch_bounds__(WAVES * 32)
void gemm_nt_wmma_bf16(const bf16* __restrict__ A1, int lda1,
                       const bf16* __restrict__ A2, int lda2, int K1, int K,
                       const bf16* __restrict__ Bw, int ldb,
                       float* __restrict__ outF, bf16* __restrict__ outB, int ldo,
                       const float* __restrict__ bias, int Nvalid)
{
    constexpr int BMt     = WAVES * 16;
    constexpr int BNt     = NTILES * 16;
    constexpr int THREADS = WAVES * 32;
    constexpr int CHUNKS  = BNt * 4;             // 16B chunks per 32-K B tile
    constexpr int CPT     = CHUNKS / THREADS;    // chunks per thread
    static_assert(CHUNKS % THREADS == 0, "copy partition");

    __shared__ bf16 lb[2][BNt * LDSROW];

    const int tid   = threadIdx.x;
    const int lane  = tid & 31;
    const int wave  = tid >> 5;
    const int nloc  = lane & 15;
    const int khalf = lane >> 4;
    const int mrow  = blockIdx.x * BMt + wave * 16 + nloc;  // A row for this lane
    const int nbase = blockIdx.y * BNt;

    v8f z = {0.f,0.f,0.f,0.f,0.f,0.f,0.f,0.f};
    v8f acc[NTILES];
#pragma unroll
    for (int i = 0; i < NTILES; ++i) acc[i] = z;

    // Cooperative B copy: each thread moves CPT x 16B per k-step.
    auto stage = [&](int buf, int k0) {
#pragma unroll
        for (int i = 0; i < CPT; ++i) {
            int id  = tid + i * THREADS;
            int row = id >> 2, q = id & 3;
            *(v8bf*)(&lb[buf][row * LDSROW + q * 8]) =
                *(const v8bf*)(Bw + (size_t)(nbase + row) * ldb + k0 + q * 8);
        }
    };

    stage(0, 0);                                 // prologue: k-step 0

    const int nk = K / 32;
    int cur = 0;
    for (int ks = 0; ks < nk; ++ks) {
        const int k0 = ks * 32;
        __syncthreads();                         // lb[cur] ready; prior reads done

        if (ks + 1 < nk) {                       // stage next tile into lb[1-cur]
            stage(1 - cur, k0 + 32);
            __builtin_prefetch(Bw + (size_t)(nbase + (tid >> 2)) * ldb + k0 + 96,
                               0, 1);            // global_prefetch (CDNA5)
        }

        // A fragment straight from global (per-wave unique rows).
        const bf16* arow = (k0 < K1) ? (A1 + (size_t)mrow * lda1 + k0)
                                     : (A2 + (size_t)mrow * lda2 + (k0 - K1));
        Frag a;
        a.h[0] = *(const v8bf*)(arow + khalf * 8);
        a.h[1] = *(const v8bf*)(arow + 16 + khalf * 8);

        // B fragments from LDS (identical addresses across waves -> broadcast).
        const bf16* lrow = &lb[cur][0] + khalf * 8;
#pragma unroll
        for (int nt = 0; nt < NTILES; ++nt) {
            Frag b;
            b.h[0] = *(const v8bf*)(lrow + (nt * 16 + nloc) * LDSROW);
            b.h[1] = *(const v8bf*)(lrow + (nt * 16 + nloc) * LDSROW + 16);
            acc[nt] = __builtin_amdgcn_wmma_f32_16x16x32_bf16(
                false, a.v, false, b.v, (short)0, acc[nt], false, false);
        }
        cur ^= 1;
    }

    // Store C: lanes 0-15 -> M=r, lanes 16-31 -> M=r+8, N = nloc per 16-tile.
    const int rbase = khalf * 8;
#pragma unroll
    for (int nt = 0; nt < NTILES; ++nt) {
        int gcol = nbase + nt * 16 + nloc;
        if (gcol >= Nvalid) continue;
        float bv = bias ? bias[gcol] : 0.f;
#pragma unroll
        for (int r = 0; r < 8; ++r) {
            int grow = blockIdx.x * BMt + wave * 16 + rbase + r;
            float val = acc[nt][r] + bv;
            if (outF) outF[(size_t)grow * ldo + gcol] = val;
            else      outB[(size_t)grow * ldo + gcol] = (bf16)val;
        }
    }
}

// gates[b, 0:4H] -> i,f,g,o ; update c ; h = o*tanh(c) written as bf16 to the
// recurrent state and to the layer-output sequence buffer.
__global__ void lstm_pointwise(const float* __restrict__ gates, int ldg, int Hs,
                               float* __restrict__ c,
                               bf16* __restrict__ hb, int ldh,
                               bf16* __restrict__ y, int ldy, size_t n)
{
    size_t idx = (size_t)blockIdx.x * blockDim.x + threadIdx.x;
    if (idx >= n) return;
    int b = (int)(idx / Hs), j = (int)(idx % Hs);
    const float* g = gates + (size_t)b * ldg;
    float gi = g[j], gf = g[Hs + j], gg = g[2 * Hs + j], go = g[3 * Hs + j];
    float i = 1.f / (1.f + __expf(-gi));
    float f = 1.f / (1.f + __expf(-gf));
    float t = tanhf(gg);
    float o = 1.f / (1.f + __expf(-go));
    float cc = f * c[idx] + i * t;
    c[idx] = cc;
    float h = o * tanhf(cc);
    hb[(size_t)b * ldh + j] = (bf16)h;
    y[(size_t)b * ldy + j] = (bf16)h;
}

// f32 [N,K] -> bf16 [Npad, ldd] at column offset, zero-padded.
__global__ void convert_pad_bf16(bf16* __restrict__ dst, int ldd, int coloff,
                                 const float* __restrict__ src,
                                 int N, int K, int Kpad, size_t n)
{
    size_t idx = (size_t)blockIdx.x * blockDim.x + threadIdx.x;
    if (idx >= n) return;
    int r = (int)(idx / Kpad), k = (int)(idx % Kpad);
    float v = (r < N && k < K) ? src[(size_t)r * K + k] : 0.f;
    dst[(size_t)r * ldd + coloff + k] = (bf16)v;
}

__global__ void bias_combine(float* __restrict__ dst, const float* __restrict__ b1,
                             const float* __restrict__ b2, int N, int Npad)
{
    int i = blockIdx.x * blockDim.x + threadIdx.x;
    if (i >= Npad) return;
    dst[i] = (i < N) ? (b1[i] + b2[i]) : 0.f;
}

__global__ void init_state(const float* __restrict__ h0, const float* __restrict__ c0,
                           bf16* __restrict__ hb, float* __restrict__ c,
                           int Hs, int Hpad, size_t n)
{
    size_t idx = (size_t)blockIdx.x * blockDim.x + threadIdx.x;
    if (idx >= n) return;
    int b = (int)(idx / Hpad), j = (int)(idx % Hpad);
    float hv = (j < Hs) ? h0[(size_t)b * Hs + j] : 0.f;
    hb[idx] = (bf16)hv;
    if (j < Hs) c[(size_t)b * Hs + j] = c0[(size_t)b * Hs + j];
}

__global__ void embed_kernel(const int* __restrict__ tokens, const float* __restrict__ emb,
                             bf16* __restrict__ out, size_t n)
{
    size_t idx = (size_t)blockIdx.x * blockDim.x + threadIdx.x;
    if (idx >= n) return;
    int m = (int)(idx / EP), k = (int)(idx % EP);
    int tok = tokens[m];
    float v = (k < EE) ? emb[(size_t)tok * EE + k] : 0.f;
    out[idx] = (bf16)v;
}

__global__ void zero_bf16_kernel(bf16* __restrict__ p, size_t n)
{
    size_t idx = (size_t)blockIdx.x * blockDim.x + threadIdx.x;
    if (idx < n) p[idx] = (bf16)0.f;
}

// ---------------------------------------------------------------------------
extern "C" void kernel_launch(void* const* d_in, const int* in_sizes, int n_in,
                              void* d_out, int out_size, void* d_ws, size_t ws_size,
                              hipStream_t stream)
{
    const int*   tokens = (const int*)  d_in[0];
    const float* h0_0   = (const float*)d_in[1];
    const float* c0_0   = (const float*)d_in[2];
    const float* h0_1   = (const float*)d_in[3];
    const float* c0_1   = (const float*)d_in[4];
    const float* h0_2   = (const float*)d_in[5];
    const float* c0_2   = (const float*)d_in[6];
    const float* embw   = (const float*)d_in[7];
    const float* W_ih0  = (const float*)d_in[8];
    const float* W_hh0  = (const float*)d_in[9];
    const float* b_ih0  = (const float*)d_in[10];
    const float* b_hh0  = (const float*)d_in[11];
    const float* W_ih1  = (const float*)d_in[12];
    const float* W_hh1  = (const float*)d_in[13];
    const float* b_ih1  = (const float*)d_in[14];
    const float* b_hh1  = (const float*)d_in[15];
    const float* W_ih2  = (const float*)d_in[16];
    const float* W_hh2  = (const float*)d_in[17];
    const float* b_ih2  = (const float*)d_in[18];
    const float* b_hh2  = (const float*)d_in[19];
    const float* dec_b  = (const float*)d_in[20];
    float* out = (float*)d_out;

    char* base = (char*)d_ws;
    size_t off = 0;
    auto alloc = [&](size_t bytes) -> void* {
        void* p = base + off;
        off = (off + bytes + 255) & ~(size_t)255;
        return p;
    };

    bf16*  emb_b = (bf16*) alloc((size_t)VP * EP * 2);            // tied decoder weight
    bf16*  Wc0   = (bf16*) alloc((size_t)G0P * (EP + HP) * 2);    // [Wih0;Whh0]
    bf16*  Wc1   = (bf16*) alloc((size_t)G0P * (HP + HP) * 2);
    bf16*  Wc2   = (bf16*) alloc((size_t)G2P * (HP + EP) * 2);
    float* bias0 = (float*)alloc((size_t)G0P * 4);
    float* bias1 = (float*)alloc((size_t)G0P * 4);
    float* bias2 = (float*)alloc((size_t)G2P * 4);
    bf16*  xe    = (bf16*) alloc((size_t)MR * EP * 2);            // embedded input
    bf16*  y0    = (bf16*) alloc((size_t)MR * HP * 2);
    bf16*  y1    = (bf16*) alloc((size_t)MR * HP * 2);
    bf16*  y2    = (bf16*) alloc((size_t)MR * EP * 2);
    float* gates = (float*)alloc((size_t)BB * G0P * 4);
    bf16*  hb0   = (bf16*) alloc((size_t)BB * HP * 2);
    bf16*  hb1   = (bf16*) alloc((size_t)BB * HP * 2);
    bf16*  hb2   = (bf16*) alloc((size_t)BB * EP * 2);
    float* cs0   = (float*)alloc((size_t)BB * HH * 4);
    float* cs1   = (float*)alloc((size_t)BB * HH * 4);
    float* cs2   = (float*)alloc((size_t)BB * EE * 4);
    (void)ws_size; (void)in_sizes; (void)n_in; (void)out_size;

    auto cvt = [&](bf16* dst, int ldd, int coloff, const float* src,
                   int N, int K, int Npad, int Kpad) {
        size_t n = (size_t)Npad * Kpad;
        convert_pad_bf16<<<(unsigned)((n + 255) / 256), 256, 0, stream>>>(
            dst, ldd, coloff, src, N, K, Kpad, n);
    };

    // Weight / embedding conversion to padded bf16 (every call; deterministic).
    cvt(emb_b, EP, 0, embw, VV, EE, VP, EP);
    cvt(Wc0, EP + HP, 0,  W_ih0, 4 * HH, EE, G0P, EP);
    cvt(Wc0, EP + HP, EP, W_hh0, 4 * HH, HH, G0P, HP);
    cvt(Wc1, 2 * HP, 0,   W_ih1, 4 * HH, HH, G0P, HP);
    cvt(Wc1, 2 * HP, HP,  W_hh1, 4 * HH, HH, G0P, HP);
    cvt(Wc2, HP + EP, 0,  W_ih2, 4 * EE, HH, G2P, HP);
    cvt(Wc2, HP + EP, HP, W_hh2, 4 * EE, EE, G2P, EP);
    bias_combine<<<(G0P + 255) / 256, 256, 0, stream>>>(bias0, b_ih0, b_hh0, 4 * HH, G0P);
    bias_combine<<<(G0P + 255) / 256, 256, 0, stream>>>(bias1, b_ih1, b_hh1, 4 * HH, G0P);
    bias_combine<<<(G2P + 255) / 256, 256, 0, stream>>>(bias2, b_ih2, b_hh2, 4 * EE, G2P);

    { size_t n = (size_t)MR * EP;
      embed_kernel<<<(unsigned)((n + 255) / 256), 256, 0, stream>>>(tokens, embw, xe, n); }
    { size_t n = (size_t)MR * HP;
      zero_bf16_kernel<<<(unsigned)((n + 255) / 256), 256, 0, stream>>>(y0, n);
      zero_bf16_kernel<<<(unsigned)((n + 255) / 256), 256, 0, stream>>>(y1, n); }
    { size_t n = (size_t)MR * EP;
      zero_bf16_kernel<<<(unsigned)((n + 255) / 256), 256, 0, stream>>>(y2, n); }

    // One LSTM layer: 70 sequential (fused-GEMM -> pointwise) steps.
    // Skinny <2,4> tiles: grid (4, N/64) = 288 blocks/step for better spread
    // of the serial critical path across WGPs.
    auto run_layer = [&](const bf16* xsrc, int ldx, const bf16* Wc, const float* bias,
                         int N4pad, int Hs, int Hpad, bf16* hb, float* cst, bf16* yout,
                         const float* h0, const float* c0) {
        { size_t n = (size_t)BB * Hpad;
          init_state<<<(unsigned)((n + 255) / 256), 256, 0, stream>>>(h0, c0, hb, cst, Hs, Hpad, n); }
        const int K1 = ldx, K = ldx + Hpad;
        dim3 grid(BB / 32, N4pad / 64);
        size_t pwn = (size_t)BB * Hs;
        unsigned pwb = (unsigned)((pwn + 255) / 256);
        for (int t = 0; t < TT; ++t) {
            gemm_nt_wmma_bf16<2, 4><<<grid, 64, 0, stream>>>(
                xsrc + (size_t)t * BB * ldx, ldx, hb, Hpad, K1, K,
                Wc, K, gates, nullptr, N4pad, bias, N4pad);
            lstm_pointwise<<<pwb, 256, 0, stream>>>(
                gates, N4pad, Hs, cst, hb, Hpad,
                yout + (size_t)t * BB * Hpad, Hpad, pwn);
        }
    };

    run_layer(xe, EP, Wc0, bias0, G0P, HH, HP, hb0, cs0, y0, h0_0, c0_0);
    run_layer(y0, HP, Wc1, bias1, G0P, HH, HP, hb1, cs1, y1, h0_1, c0_1);
    run_layer(y1, HP, Wc2, bias2, G2P, EE, EP, hb2, cs2, y2, h0_2, c0_2);

    // Tied decoder: logits = y2 @ emb^T + dec_b   (dominant GEMM, ~239 GF).
    // Fat <8,8> tiles: 128x128 blocks, grid (70, 260).
    dim3 gd(MR / 128, VP / 128);
    gemm_nt_wmma_bf16<8, 8><<<gd, 256, 0, stream>>>(
        y2, EP, nullptr, 0, EP, EP, emb_b, EP,
        out, nullptr, VV, dec_b, VV);
}
